// MatMamba2_24180665876838
// MI455X (gfx1250) — compile-verified, matrix operand order checked by reading, not run on hardware
//
#include <hip/hip_runtime.h>
#include <hip/hip_bf16.h>

typedef __attribute__((ext_vector_type(8)))  float   v8f;
typedef __attribute__((ext_vector_type(16))) __bf16  v16bf;
typedef __attribute__((ext_vector_type(4)))  __bf16  v4bf;
typedef __attribute__((ext_vector_type(4)))  int     v4i;

#define LSEQ    4096
#define DMODEL  2048
#define DINNER  4096
#define DSTATE  128
#define CONVDIM 4352
#define DINPROJ 8512
#define NH      64
#define HD      64
#define NCHUNK  16
#define CC      256

#if __has_builtin(__builtin_amdgcn_global_load_async_to_lds_b128)
#define USE_ASYNC 1
#else
#define USE_ASYNC 0
#endif

// ---------------------------------------------------------------------------
// WMMA helpers
// ---------------------------------------------------------------------------
__device__ __forceinline__ v8f wmma_bf(v16bf a, v16bf b, v8f c) {
  return __builtin_amdgcn_wmma_f32_16x16x32_bf16(
      false, a, false, b, (short)0, c, false, false);
}

// 16x32 bf16 fragment from LDS tile (row stride 32), ISA 7.12.2 layout.
__device__ __forceinline__ v16bf ld_frag(const __bf16* s, int rowBase, int lane) {
  const __bf16* p = s + (rowBase + (lane & 15)) * 32 + ((lane & 16) ? 8 : 0);
  v16bf f;
#pragma unroll
  for (int i = 0; i < 8; ++i) { f[i] = p[i]; f[i + 8] = p[i + 16]; }
  return f;
}

__device__ __forceinline__ float sigmoidf_(float x) { return 1.f / (1.f + __expf(-x)); }
__device__ __forceinline__ float siluf_(float x)    { return x * sigmoidf_(x); }
__device__ __forceinline__ float softplusf_(float x){ return (x > 20.f) ? x : log1pf(__expf(x)); }

__device__ __forceinline__ v4bf cvt4(float4 v) {
  v4bf r; r[0] = (__bf16)v.x; r[1] = (__bf16)v.y; r[2] = (__bf16)v.z; r[3] = (__bf16)v.w;
  return r;
}
__device__ __forceinline__ v4bf cvt4s(float4 v, float s) {
  v4bf r; r[0] = (__bf16)(v.x*s); r[1] = (__bf16)(v.y*s); r[2] = (__bf16)(v.z*s); r[3] = (__bf16)(v.w*s);
  return r;
}

#if USE_ASYNC
__device__ __forceinline__ void async_cp16(const __bf16* g, __bf16* l) {
  __builtin_amdgcn_global_load_async_to_lds_b128(
      (__attribute__((address_space(1))) v4i*)(g),
      (__attribute__((address_space(3))) v4i*)(l), 0, 0);
}
__device__ __forceinline__ void wait_async0() {
#if __has_builtin(__builtin_amdgcn_s_wait_asynccnt)
  __builtin_amdgcn_s_wait_asynccnt(0);
#else
  asm volatile("s_wait_asynccnt 0x0" ::: "memory");
#endif
}
#endif

// ---------------------------------------------------------------------------
// Kernel 0: fp32 -> bf16 conversion (one-time per operand)
// ---------------------------------------------------------------------------
__global__ void __launch_bounds__(256)
cvt_bf16_kernel(const float* __restrict__ src, __bf16* __restrict__ dst, long n) {
  long i4 = ((long)blockIdx.x * 256 + threadIdx.x) * 4;
  if (i4 < n) *(v4bf*)(dst + i4) = cvt4(*(const float4*)(src + i4));
}

// ---------------------------------------------------------------------------
// Kernel 1: batched GEMM  C[M,N] = A[M,K] * B[N,K]^T, bf16 in / fp32 out.
// 128x128 tile, BK=32, 256 thr (8 waves 4x2, wave = 32x64 = 2x4 wmma tiles).
// Double-buffered LDS; staging = raw 16B copies (async-to-LDS on CDNA5).
// Edge rows are clamped, never zeroed: clamped B rows only produce garbage in
// C columns >= N, which the store guard discards (K, M always tile-aligned).
// ---------------------------------------------------------------------------
__global__ void __launch_bounds__(256)
gemm_bf16_wmma(const __bf16* __restrict__ A, const __bf16* __restrict__ B,
               float* __restrict__ C, int M, int N, int K,
               int lda, int ldb, int ldc,
               long sA, long sB, long sC) {
  A += (long)blockIdx.z * sA;
  B += (long)blockIdx.z * sB;
  C += (long)blockIdx.z * sC;

  __shared__ __align__(16) __bf16 As[2][128 * 32];
  __shared__ __align__(16) __bf16 Bs[2][128 * 32];

  const int tid  = threadIdx.x;
  const int lane = tid & 31;
  const int wave = tid >> 5;
  const int wm   = wave >> 1;   // 0..3 (M)
  const int wn   = wave & 1;    // 0..1 (N)
  const int m0   = blockIdx.x * 128;
  const int n0   = blockIdx.y * 128;

  // staging geometry: 2 chunks of 8 bf16 (16 B) per thread per matrix
  long offA[2], offB[2];
  int dstE[2];
#pragma unroll
  for (int i = 0; i < 2; ++i) {
    int elt = (i * 256 + tid) * 8;          // tile-linear bf16 element, 0..4088
    int r = elt >> 5, kc = elt & 31;
    int ga = m0 + r; if (ga >= M) ga = M - 1;
    int gb = n0 + r; if (gb >= N) gb = N - 1;
    offA[i] = (long)ga * lda + kc;
    offB[i] = (long)gb * ldb + kc;
    dstE[i] = elt;
  }

  auto stage = [&](int k0, int b) {
#pragma unroll
    for (int i = 0; i < 2; ++i) {
#if USE_ASYNC
      async_cp16(A + offA[i] + k0, &As[b][dstE[i]]);
      async_cp16(B + offB[i] + k0, &Bs[b][dstE[i]]);
#else
      *(uint4*)(&As[b][dstE[i]]) = *(const uint4*)(A + offA[i] + k0);
      *(uint4*)(&Bs[b][dstE[i]]) = *(const uint4*)(B + offB[i] + k0);
#endif
    }
  };

  v8f acc[2][4];
#pragma unroll
  for (int i = 0; i < 2; ++i)
#pragma unroll
    for (int j = 0; j < 4; ++j) acc[i][j] = v8f{};

  const int NT = K >> 5;
  stage(0, 0);
#if USE_ASYNC
  wait_async0();
#endif
  __syncthreads();

  for (int t = 0; t < NT; ++t) {
    const int cur = t & 1;
    if (t + 1 < NT) stage((t + 1) << 5, cur ^ 1);

    v16bf af[2], bfg[4];
#pragma unroll
    for (int i = 0; i < 2; ++i) af[i]  = ld_frag(&As[cur][0], wm * 32 + i * 16, lane);
#pragma unroll
    for (int j = 0; j < 4; ++j) bfg[j] = ld_frag(&Bs[cur][0], wn * 64 + j * 16, lane);
#pragma unroll
    for (int i = 0; i < 2; ++i)
#pragma unroll
      for (int j = 0; j < 4; ++j) acc[i][j] = wmma_bf(af[i], bfg[j], acc[i][j]);

#if USE_ASYNC
    wait_async0();                 // next buffer fully landed in LDS
#endif
    __syncthreads();
  }

#pragma unroll
  for (int i = 0; i < 2; ++i)
#pragma unroll
    for (int j = 0; j < 4; ++j) {
      int tm = m0 + wm * 32 + i * 16 + ((lane & 16) ? 8 : 0);
      int tn = n0 + wn * 64 + j * 16 + (lane & 15);
      if (tn < N) {
#pragma unroll
        for (int r = 0; r < 8; ++r) {
          int row = tm + r;
          if (row < M) C[(long)row * ldc + tn] = acc[i][j][r];
        }
      }
    }
}

// ---------------------------------------------------------------------------
// Kernel 2: causal depthwise conv1d + bias + SiLU; writes fp32 + bf16 copies.
// ---------------------------------------------------------------------------
__global__ void __launch_bounds__(256)
conv_silu_kernel(const float* __restrict__ Z, const float* __restrict__ CW,
                 const float* __restrict__ CB, float* __restrict__ XBC,
                 __bf16* __restrict__ XBCb) {
  long idx4 = ((long)blockIdx.x * 256 + threadIdx.x) * 4;
  int l = (int)(idx4 / CONVDIM);
  int c = (int)(idx4 - (long)l * CONVDIM);
  float4 s = *(const float4*)(CB + c);
#pragma unroll
  for (int j = 0; j < 4; ++j) {
    int t = l - 3 + j;
    if (t >= 0) {
      float4 zv = *(const float4*)(Z + (long)t * DINPROJ + DINNER + c);
      s.x += CW[(c + 0) * 4 + j] * zv.x;
      s.y += CW[(c + 1) * 4 + j] * zv.y;
      s.z += CW[(c + 2) * 4 + j] * zv.z;
      s.w += CW[(c + 3) * 4 + j] * zv.w;
    }
  }
  float4 o;
  o.x = siluf_(s.x); o.y = siluf_(s.y); o.z = siluf_(s.z); o.w = siluf_(s.w);
  *(float4*)(XBC + idx4) = o;
  *(v4bf*)(XBCb + idx4)  = cvt4(o);
}

// ---------------------------------------------------------------------------
// Kernel 3: dt = softplus(dt_raw + dt_bias); dAcum = cumsum(dt*A) per chunk.
// ---------------------------------------------------------------------------
__global__ void __launch_bounds__(256)
dt_dacum_kernel(const float* __restrict__ Z, const float* __restrict__ dt_bias,
                const float* __restrict__ A_log,
                float* __restrict__ DT, float* __restrict__ DAC) {
  int zh = blockIdx.x;
  int z = zh >> 6, h = zh & 63;
  int c = threadIdx.x;
  int l = z * CC + c;
  float dtv = softplusf_(Z[(long)l * DINPROJ + (DINNER + CONVDIM) + h] + dt_bias[h]);
  DT[(long)l * NH + h] = dtv;
  float A = -__expf(A_log[h]);
  __shared__ float s[CC];
  s[c] = dtv * A;
  __syncthreads();
  for (int off = 1; off < CC; off <<= 1) {
    float v = (c >= off) ? s[c - off] : 0.f;
    __syncthreads();
    s[c] += v;
    __syncthreads();
  }
  DAC[(long)zh * CC + c] = s[c];
}

// ---------------------------------------------------------------------------
// Kernel 4: per (chunk, head): states[p,n] = sum_c Xdt[c,p]*decay[c]*B_c[c,n]
// ---------------------------------------------------------------------------
__global__ void __launch_bounds__(256)
states_kernel(const float* __restrict__ XBC, const float* __restrict__ DT,
              const float* __restrict__ DAC, float* __restrict__ STATES) {
  int zh = blockIdx.x;
  int z = zh >> 6, h = zh & 63;
  const int tid = threadIdx.x, lane = tid & 31, wave = tid >> 5;
  const int wm = wave >> 2, wn = wave & 3;

  __shared__ __align__(16) __bf16 As[64 * 32];
  __shared__ __align__(16) __bf16 Bs[128 * 32];

  float dAlast = DAC[(long)zh * CC + (CC - 1)];

  v8f acc[2][2];
#pragma unroll
  for (int i = 0; i < 2; ++i)
#pragma unroll
    for (int j = 0; j < 2; ++j) acc[i][j] = v8f{};

  for (int c0 = 0; c0 < CC; c0 += 32) {
    float4 xa[2]; float sca[2]; int pA[2], cA[2];
#pragma unroll
    for (int i = 0; i < 2; ++i) {
      int idx4 = (i * 256 + tid) * 4;
      pA[i] = idx4 & 63; cA[i] = idx4 >> 6;
      int l = z * CC + c0 + cA[i];
      xa[i] = *(const float4*)(XBC + (long)l * CONVDIM + h * HD + pA[i]);
      sca[i] = DT[(long)l * NH + h] * __expf(dAlast - DAC[(long)zh * CC + c0 + cA[i]]);
    }
    float4 xb[4]; int nB[4], cB[4];
#pragma unroll
    for (int i = 0; i < 4; ++i) {
      int idx4 = (i * 256 + tid) * 4;
      nB[i] = idx4 & 127; cB[i] = idx4 >> 7;
      int l = z * CC + c0 + cB[i];
      xb[i] = *(const float4*)(XBC + (long)l * CONVDIM + DINNER + nB[i]);
    }
#pragma unroll
    for (int i = 0; i < 2; ++i) {
      As[(pA[i] + 0) * 32 + cA[i]] = (__bf16)(xa[i].x * sca[i]);
      As[(pA[i] + 1) * 32 + cA[i]] = (__bf16)(xa[i].y * sca[i]);
      As[(pA[i] + 2) * 32 + cA[i]] = (__bf16)(xa[i].z * sca[i]);
      As[(pA[i] + 3) * 32 + cA[i]] = (__bf16)(xa[i].w * sca[i]);
    }
#pragma unroll
    for (int i = 0; i < 4; ++i) {
      Bs[(nB[i] + 0) * 32 + cB[i]] = (__bf16)xb[i].x;
      Bs[(nB[i] + 1) * 32 + cB[i]] = (__bf16)xb[i].y;
      Bs[(nB[i] + 2) * 32 + cB[i]] = (__bf16)xb[i].z;
      Bs[(nB[i] + 3) * 32 + cB[i]] = (__bf16)xb[i].w;
    }
    __syncthreads();
    v16bf af[2], bfg[2];
#pragma unroll
    for (int i = 0; i < 2; ++i) af[i]  = ld_frag(As, wm * 32 + i * 16, lane);
#pragma unroll
    for (int j = 0; j < 2; ++j) bfg[j] = ld_frag(Bs, wn * 32 + j * 16, lane);
#pragma unroll
    for (int i = 0; i < 2; ++i)
#pragma unroll
      for (int j = 0; j < 2; ++j) acc[i][j] = wmma_bf(af[i], bfg[j], acc[i][j]);
    __syncthreads();
  }

#pragma unroll
  for (int i = 0; i < 2; ++i)
#pragma unroll
    for (int j = 0; j < 2; ++j) {
      int tm = wm * 32 + i * 16 + ((lane & 16) ? 8 : 0);
      int tn = wn * 32 + j * 16 + (lane & 15);
#pragma unroll
      for (int r = 0; r < 8; ++r)
        STATES[((long)zh * HD + (tm + r)) * DSTATE + tn] = acc[i][j][r];
    }
}

// ---------------------------------------------------------------------------
// Kernel 5: sequential inter-chunk recurrence (float4).
// ---------------------------------------------------------------------------
__global__ void __launch_bounds__(256)
prev_scan_kernel(const float* __restrict__ STATES, const float* __restrict__ DAC,
                 float* __restrict__ PREV) {
  int h = blockIdx.x, tid = threadIdx.x;
  __shared__ float dec[NCHUNK];
  if (tid < NCHUNK) dec[tid] = __expf(DAC[(long)(tid * NH + h) * CC + (CC - 1)]);
  __syncthreads();
#pragma unroll 2
  for (int e = 0; e < 8; ++e) {
    int idx4 = (e * 256 + tid) * 4;
    float4 run = {0.f, 0.f, 0.f, 0.f};
    for (int z = 0; z < NCHUNK; ++z) {
      long o = (long)(z * NH + h) * (HD * DSTATE) + idx4;
      *(float4*)(PREV + o) = run;
      float4 st = *(const float4*)(STATES + o);
      float d = dec[z];
      run.x = run.x * d + st.x; run.y = run.y * d + st.y;
      run.z = run.z * d + st.z; run.w = run.w * d + st.w;
    }
  }
}

// ---------------------------------------------------------------------------
// Kernel 6: per (chunk, head): Y = (L∘CB)@Xdt + (C∘e^dAcum)@prev + x*D[h]
// ---------------------------------------------------------------------------
__global__ void __launch_bounds__(256)
ydiag_kernel(const float* __restrict__ XBC, const float* __restrict__ DT,
             const float* __restrict__ DAC, const float* __restrict__ CBm,
             const float* __restrict__ PREV, const float* __restrict__ Dv,
             float* __restrict__ Y) {
  int zh = blockIdx.x;
  int z = zh >> 6, h = zh & 63;
  const int tid = threadIdx.x, lane = tid & 31, wave = tid >> 5;

  __shared__ __align__(16) __bf16 As[256 * 32];
  __shared__ __align__(16) __bf16 Bs[64 * 32];
  __shared__ float dac[CC];

  dac[tid] = DAC[(long)zh * CC + tid];
  __syncthreads();

  v8f acc[2][4];
#pragma unroll
  for (int i = 0; i < 2; ++i)
#pragma unroll
    for (int j = 0; j < 4; ++j) acc[i][j] = v8f{};

  for (int s0 = 0; s0 < CC; s0 += 32) {
#pragma unroll
    for (int i = 0; i < 8; ++i) {
      int idx4 = (i * 256 + tid) * 4;
      int c = idx4 >> 5, ss = idx4 & 31;
      int sg = s0 + ss;
      float4 v = *(const float4*)(CBm + (long)z * CC * CC + (long)c * CC + sg);
      float dc = dac[c];
      float e0 = (sg + 0 <= c) ? __expf(dc - dac[sg + 0]) : 0.f;
      float e1 = (sg + 1 <= c) ? __expf(dc - dac[sg + 1]) : 0.f;
      float e2 = (sg + 2 <= c) ? __expf(dc - dac[sg + 2]) : 0.f;
      float e3 = (sg + 3 <= c) ? __expf(dc - dac[sg + 3]) : 0.f;
      v.x *= e0; v.y *= e1; v.z *= e2; v.w *= e3;
      *(v4bf*)(As + idx4) = cvt4(v);
    }
    float4 xb[2]; float dtb[2]; int pB[2], sB[2];
#pragma unroll
    for (int i = 0; i < 2; ++i) {
      int idx4 = (i * 256 + tid) * 4;
      pB[i] = idx4 & 63; sB[i] = idx4 >> 6;
      int l = z * CC + s0 + sB[i];
      xb[i] = *(const float4*)(XBC + (long)l * CONVDIM + h * HD + pB[i]);
      dtb[i] = DT[(long)l * NH + h];
    }
#pragma unroll
    for (int i = 0; i < 2; ++i) {
      Bs[(pB[i] + 0) * 32 + sB[i]] = (__bf16)(xb[i].x * dtb[i]);
      Bs[(pB[i] + 1) * 32 + sB[i]] = (__bf16)(xb[i].y * dtb[i]);
      Bs[(pB[i] + 2) * 32 + sB[i]] = (__bf16)(xb[i].z * dtb[i]);
      Bs[(pB[i] + 3) * 32 + sB[i]] = (__bf16)(xb[i].w * dtb[i]);
    }
    __syncthreads();
    v16bf af[2], bfg[4];
#pragma unroll
    for (int i = 0; i < 2; ++i) af[i]  = ld_frag(As, wave * 32 + i * 16, lane);
#pragma unroll
    for (int j = 0; j < 4; ++j) bfg[j] = ld_frag(Bs, j * 16, lane);
#pragma unroll
    for (int i = 0; i < 2; ++i)
#pragma unroll
      for (int j = 0; j < 4; ++j) acc[i][j] = wmma_bf(af[i], bfg[j], acc[i][j]);
    __syncthreads();
  }

  for (int n0 = 0; n0 < DSTATE; n0 += 32) {
#pragma unroll
    for (int i = 0; i < 8; ++i) {
      int idx4 = (i * 256 + tid) * 4;
      int c = idx4 >> 5, nn = idx4 & 31;
      float4 v = *(const float4*)(XBC + (long)(z * CC + c) * CONVDIM +
                                  DINNER + DSTATE + n0 + nn);
      *(v4bf*)(As + idx4) = cvt4s(v, __expf(dac[c]));
    }
#pragma unroll
    for (int i = 0; i < 2; ++i) {
      int idx4 = (i * 256 + tid) * 4;          // == p*32+nn
      int p = idx4 >> 5, nn = idx4 & 31;
      float4 v = *(const float4*)(PREV + ((long)zh * HD + p) * DSTATE + n0 + nn);
      *(v4bf*)(Bs + idx4) = cvt4(v);
    }
    __syncthreads();
    v16bf af[2], bfg[4];
#pragma unroll
    for (int i = 0; i < 2; ++i) af[i]  = ld_frag(As, wave * 32 + i * 16, lane);
#pragma unroll
    for (int j = 0; j < 4; ++j) bfg[j] = ld_frag(Bs, j * 16, lane);
#pragma unroll
    for (int i = 0; i < 2; ++i)
#pragma unroll
      for (int j = 0; j < 4; ++j) acc[i][j] = wmma_bf(af[i], bfg[j], acc[i][j]);
    __syncthreads();
  }

  float Dh = Dv[h];
#pragma unroll
  for (int i = 0; i < 2; ++i)
#pragma unroll
    for (int j = 0; j < 4; ++j) {
      int tm = wave * 32 + i * 16 + ((lane & 16) ? 8 : 0);
      int tn = j * 16 + (lane & 15);
#pragma unroll
      for (int r = 0; r < 8; ++r) {
        int l = z * CC + tm + r;
        int col = h * HD + tn;
        float xv = XBC[(long)l * CONVDIM + col];
        Y[(long)l * DINNER + col] = acc[i][j][r] + xv * Dh;
      }
    }
}

// ---------------------------------------------------------------------------
// Kernel 7: gated RMSNorm; reads fp32 Y, writes bf16 Y for the out_proj GEMM.
// ---------------------------------------------------------------------------
__global__ void __launch_bounds__(256)
rmsnorm_gate_kernel(float* __restrict__ Y, const float* __restrict__ Z,
                    const float* __restrict__ W, __bf16* __restrict__ Yb) {
  int l = blockIdx.x, tid = threadIdx.x;
  __shared__ float red[256];
  float ss = 0.f;
  for (int i4 = tid * 4; i4 < DINNER; i4 += 1024) {
    float4 zv = *(const float4*)(Z + (long)l * DINPROJ + i4);
    float4 yv = *(const float4*)(Y + (long)l * DINNER + i4);
    yv.x *= siluf_(zv.x); yv.y *= siluf_(zv.y);
    yv.z *= siluf_(zv.z); yv.w *= siluf_(zv.w);
    *(float4*)(Y + (long)l * DINNER + i4) = yv;
    ss += yv.x * yv.x + yv.y * yv.y + yv.z * yv.z + yv.w * yv.w;
  }
  red[tid] = ss;
  __syncthreads();
  for (int o = 128; o > 0; o >>= 1) {
    if (tid < o) red[tid] += red[tid + o];
    __syncthreads();
  }
  float rs = rsqrtf(red[0] / (float)DINNER + 1e-5f);
  for (int i4 = tid * 4; i4 < DINNER; i4 += 1024) {
    float4 yv = *(const float4*)(Y + (long)l * DINNER + i4);
    float4 wv = *(const float4*)(W + i4);
    yv.x *= rs * wv.x; yv.y *= rs * wv.y; yv.z *= rs * wv.z; yv.w *= rs * wv.w;
    *(v4bf*)(Yb + (long)l * DINNER + i4) = cvt4(yv);
  }
}

// ---------------------------------------------------------------------------
// Host-side launch
// ---------------------------------------------------------------------------
extern "C" void kernel_launch(void* const* d_in, const int* in_sizes, int n_in,
                              void* d_out, int out_size, void* d_ws, size_t ws_size,
                              hipStream_t stream) {
  (void)in_sizes; (void)n_in; (void)out_size; (void)ws_size;
  const float* u        = (const float*)d_in[0];
  const float* in_w     = (const float*)d_in[1];
  const float* conv_w   = (const float*)d_in[2];
  const float* conv_b   = (const float*)d_in[3];
  const float* dt_bias  = (const float*)d_in[4];
  const float* A_log    = (const float*)d_in[5];
  const float* Dv       = (const float*)d_in[6];
  const float* norm_w   = (const float*)d_in[7];
  const float* out_w    = (const float*)d_in[8];
  float* out = (float*)d_out;

  char* ws = (char*)d_ws;
  size_t off = 0;
  float* Z      = (float*)(ws + off); off += (size_t)LSEQ * DINPROJ * 4;
  float* XBC    = (float*)(ws + off); off += (size_t)LSEQ * CONVDIM * 4;
  float* DT     = (float*)(ws + off); off += (size_t)LSEQ * NH * 4;
  float* DAC    = (float*)(ws + off); off += (size_t)NCHUNK * NH * CC * 4;
  float* CBm    = (float*)(ws + off); off += (size_t)NCHUNK * CC * CC * 4;
  float* STATES = (float*)(ws + off); off += (size_t)NCHUNK * NH * HD * DSTATE * 4;
  float* PREV   = (float*)(ws + off); off += (size_t)NCHUNK * NH * HD * DSTATE * 4;
  float* Y      = (float*)(ws + off); off += (size_t)LSEQ * DINNER * 4;
  __bf16* Ub    = (__bf16*)(ws + off); off += (size_t)LSEQ * DMODEL * 2;
  __bf16* WIb   = (__bf16*)(ws + off); off += (size_t)DINPROJ * DMODEL * 2;
  __bf16* WOb   = (__bf16*)(ws + off); off += (size_t)DMODEL * DINNER * 2;
  __bf16* XBCb  = (__bf16*)(ws + off); off += (size_t)LSEQ * CONVDIM * 2;
  __bf16* Yb    = (__bf16*)(ws + off); off += (size_t)LSEQ * DINNER * 2;

  // 0. one-time bf16 conversions of GEMM operands
  {
    long n;
    n = (long)LSEQ * DMODEL;
    cvt_bf16_kernel<<<(n / 4 + 255) / 256, 256, 0, stream>>>(u, Ub, n);
    n = (long)DINPROJ * DMODEL;
    cvt_bf16_kernel<<<(n / 4 + 255) / 256, 256, 0, stream>>>(in_w, WIb, n);
    n = (long)DMODEL * DINNER;
    cvt_bf16_kernel<<<(n / 4 + 255) / 256, 256, 0, stream>>>(out_w, WOb, n);
  }
  { // 1. in_proj: Z = u @ in_w^T   (4096x8512, K=2048)
    dim3 g(LSEQ / 128, (DINPROJ + 127) / 128, 1);
    gemm_bf16_wmma<<<g, 256, 0, stream>>>(Ub, WIb, Z, LSEQ, DINPROJ, DMODEL,
                                          DMODEL, DMODEL, DINPROJ, 0, 0, 0);
  }
  // 2. conv + SiLU (fp32 + bf16 outputs)
  conv_silu_kernel<<<(LSEQ * CONVDIM / 4) / 256, 256, 0, stream>>>(Z, conv_w, conv_b,
                                                                   XBC, XBCb);
  // 3. dt & dAcum
  dt_dacum_kernel<<<NCHUNK * NH, CC, 0, stream>>>(Z, dt_bias, A_log, DT, DAC);
  { // 4. per-chunk CB = C_c @ B_c^T  (batch 16, 256x256, K=128)
    dim3 g(CC / 128, CC / 128, NCHUNK);
    gemm_bf16_wmma<<<g, 256, 0, stream>>>(
        XBCb + DINNER + DSTATE, XBCb + DINNER, CBm, CC, CC, DSTATE,
        CONVDIM, CONVDIM, CC,
        (long)CC * CONVDIM, (long)CC * CONVDIM, (long)CC * CC);
  }
  // 5. chunk-end states (WMMA)
  states_kernel<<<NCHUNK * NH, 256, 0, stream>>>(XBC, DT, DAC, STATES);
  // 6. inter-chunk recurrence
  prev_scan_kernel<<<NH, 256, 0, stream>>>(STATES, DAC, PREV);
  // 7. Y = diag + off-diag + D-skip (WMMA)
  ydiag_kernel<<<NCHUNK * NH, 256, 0, stream>>>(XBC, DT, DAC, CBm, PREV, Dv, Y);
  // 8. gated RMSNorm -> bf16 Y
  rmsnorm_gate_kernel<<<LSEQ, 256, 0, stream>>>(Y, Z, norm_w, Yb);
  { // 9. out_proj: out = Y @ out_w^T  (4096x2048, K=4096)
    dim3 g(LSEQ / 128, DMODEL / 128, 1);
    gemm_bf16_wmma<<<g, 256, 0, stream>>>(Yb, WOb, out, LSEQ, DMODEL, DINNER,
                                          DINNER, DINNER, DMODEL, 0, 0, 0);
  }
}
